// Model_65498251264157
// MI455X (gfx1250) — matrix operand order repaired and checked by reading, not compiled, not run on hardware
//
#include <hip/hip_runtime.h>

#define NNODES 44800
#define LDOC   350
#define NDOCS  128
#define DIM    768
#define HID    256
#define CDIM   20
#define NHEADS 8
#define NLAYERS 2
#define ALPHA_  0.15f

typedef __attribute__((ext_vector_type(16))) __bf16 v16bf;
typedef __attribute__((ext_vector_type(8)))  __bf16 v8bf;
typedef __attribute__((ext_vector_type(8)))  float  v8f;

static __device__ __forceinline__ unsigned bfb(float f) {
  unsigned u = __float_as_uint(f);
  return (u + 0x7FFFu + ((u >> 16) & 1u)) >> 16;   // RNE
}
static __device__ __forceinline__ unsigned pk2(float lo, float hi) {
  return (bfb(hi) << 16) | bfb(lo);
}
static __device__ __forceinline__ __bf16 f2bf(float f) {
  unsigned short h = (unsigned short)bfb(f);
  return __builtin_bit_cast(__bf16, h);
}
// 16 consecutive bf16 via two 16-byte loads
static __device__ __forceinline__ v16bf ld16(const __bf16* p) {
  v8bf lo = __builtin_bit_cast(v8bf, *(const uint4*)p);
  v8bf hi = __builtin_bit_cast(v8bf, *(const uint4*)(p + 8));
  return __builtin_shufflevector(lo, hi, 0, 1, 2, 3, 4, 5, 6, 7, 8, 9, 10, 11, 12, 13, 14, 15);
}
// A fragment: K groups [ka..ka+7] and [ka+16..ka+23]
static __device__ __forceinline__ v16bf ld_afrag(const __bf16* row, int ka) {
  v8bf lo = __builtin_bit_cast(v8bf, *(const uint4*)(row + ka));
  v8bf hi = __builtin_bit_cast(v8bf, *(const uint4*)(row + ka + 16));
  return __builtin_shufflevector(lo, hi, 0, 1, 2, 3, 4, 5, 6, 7, 8, 9, 10, 11, 12, 13, 14, 15);
}

// ---------------------------------------------------------------------------
// one-time conversions (bf16 + TRANSPOSE so GEMM B-fragments are K-contiguous)
// ---------------------------------------------------------------------------
// W1 [768,256] f32 -> W1t [256,768] bf16 (W1t[m][k] = W1[k][m])
__global__ void k_cvt_w1t(const float* __restrict__ W1, __bf16* __restrict__ W1t) {
  int t = blockIdx.x * blockDim.x + threadIdx.x;
  if (t < DIM * HID) {
    const int k = t / HID, m = t - k * HID;
    W1t[(long)m * DIM + k] = f2bf(W1[t]);
  }
}
// W2 [256,20] f32 -> W2t [32,256] bf16 (rows 20..31 zero)
__global__ void k_cvt_w2t(const float* __restrict__ W2, __bf16* __restrict__ W2t) {
  int t = blockIdx.x * blockDim.x + threadIdx.x;
  if (t < 32 * HID) {
    const int c = t / HID, k = t - c * HID;
    W2t[t] = (c < CDIM) ? f2bf(W2[k * CDIM + c]) : f2bf(0.f);
  }
}
// gather emb[node_ids[row]] -> Xg bf16 [N,768]; fully coalesced
__global__ __launch_bounds__(192) void k_gather_cvt(
    const int* __restrict__ node_ids, const float* __restrict__ emb,
    unsigned* __restrict__ Xg) {
  const int row = blockIdx.x;
  const float4* in = (const float4*)(emb + (long)node_ids[row] * DIM);
  uint2* outp = (uint2*)(Xg + (long)row * (DIM / 2));
  const float4 v = in[threadIdx.x];
  uint2 o;
  o.x = pk2(v.x, v.y);
  o.y = pk2(v.z, v.w);
  outp[threadIdx.x] = o;
}

// ---------------------------------------------------------------------------
// GEMM1: X1 = relu(Xg @ W1) -> bf16 [N,256]
// grid = NNODES/16, block = 128 (4 waves); wave owns 16x64 (4 WMMA tiles),
// block covers 16 rows x all 256 cols. Pure b128 loads -> WMMA, no LDS.
// ---------------------------------------------------------------------------
__global__ __launch_bounds__(128) void k_gemm1(
    const __bf16* __restrict__ Xg, const __bf16* __restrict__ W1t,
    const float* __restrict__ bias1, __bf16* __restrict__ X1) {
  const int wave = threadIdx.x >> 5;
  const int lane = threadIdx.x & 31;
  const int m    = lane & 15;
  const int sel  = lane >> 4;
  const int rowTile = blockIdx.x * 16;
  const int colBase = wave * 64;

  const __bf16* arow  = Xg + (long)(rowTile + m) * DIM;
  const __bf16* brow0 = W1t + (long)(colBase + m) * DIM;
  const __bf16* brow1 = brow0 + 16 * DIM;
  const __bf16* brow2 = brow0 + 32 * DIM;
  const __bf16* brow3 = brow0 + 48 * DIM;

  v8f acc0 = {}, acc1 = {}, acc2 = {}, acc3 = {};
  for (int k0 = 0; k0 < DIM; k0 += 32) {
    __builtin_prefetch((const void*)(arow + k0 + 128), 0, 0);
    const int ka = k0 + sel * 16;   // A: lanes0-15 K=0..15, lanes16-31 K=16..31
    const int kb = k0 + sel * 16;   // B: lanes0-15 K=0..15, lanes16-31 K=16..31
    v16bf a  = ld16(arow + ka);
    v16bf b0 = ld16(brow0 + kb);
    v16bf b1 = ld16(brow1 + kb);
    v16bf b2 = ld16(brow2 + kb);
    v16bf b3 = ld16(brow3 + kb);
    acc0 = __builtin_amdgcn_wmma_f32_16x16x32_bf16(false, a, false, b0, (short)0, acc0, false, false);
    acc1 = __builtin_amdgcn_wmma_f32_16x16x32_bf16(false, a, false, b1, (short)0, acc1, false, false);
    acc2 = __builtin_amdgcn_wmma_f32_16x16x32_bf16(false, a, false, b2, (short)0, acc2, false, false);
    acc3 = __builtin_amdgcn_wmma_f32_16x16x32_bf16(false, a, false, b3, (short)0, acc3, false, false);
  }
  // D: lane holds col N = lane&15, rows M = r + 8*sel
  const v8f* accs[4] = {&acc0, &acc1, &acc2, &acc3};
#pragma unroll
  for (int t = 0; t < 4; ++t) {
    const int c = colBase + t * 16 + m;
    const float bv = bias1[c];
    const v8f& av = *accs[t];
#pragma unroll
    for (int r = 0; r < 8; ++r) {
      const int orow = rowTile + r + sel * 8;
      float v = av[r] + bv;
      v = v > 0.f ? v : 0.f;
      X1[(long)orow * HID + c] = f2bf(v);
    }
  }
}

// ---------------------------------------------------------------------------
// GEMM2: h = X1 @ W2 + b2 -> f32 [N,20] (B padded to 32 cols)
// grid = NNODES/64, block = 128; wave w -> row tile; 2 col tiles per wave
// ---------------------------------------------------------------------------
__global__ __launch_bounds__(128) void k_gemm2(
    const __bf16* __restrict__ X1, const __bf16* __restrict__ W2t,
    const float* __restrict__ b2, float* __restrict__ h) {
  const int wave = threadIdx.x >> 5;
  const int lane = threadIdx.x & 31;
  const int m    = lane & 15;
  const int sel  = lane >> 4;
  const int rowTile = (blockIdx.x * 4 + wave) * 16;

  const __bf16* arow  = X1 + (long)(rowTile + m) * HID;
  const __bf16* brow0 = W2t + (long)m * HID;
  const __bf16* brow1 = W2t + (long)(16 + m) * HID;

  v8f acc0 = {}, acc1 = {};
  for (int k0 = 0; k0 < HID; k0 += 32) {
    const int ka = k0 + sel * 16;
    const int kb = k0 + sel * 16;
    v16bf a  = ld16(arow + ka);
    v16bf b0 = ld16(brow0 + kb);
    v16bf b1 = ld16(brow1 + kb);
    acc0 = __builtin_amdgcn_wmma_f32_16x16x32_bf16(false, a, false, b0, (short)0, acc0, false, false);
    acc1 = __builtin_amdgcn_wmma_f32_16x16x32_bf16(false, a, false, b1, (short)0, acc1, false, false);
  }
  const int c0 = m, c1 = 16 + m;   // c0 always < 20; c1 only if < 20
  const float bv0 = b2[c0];
#pragma unroll
  for (int r = 0; r < 8; ++r) {
    const int orow = rowTile + r + sel * 8;
    h[(long)orow * CDIM + c0] = acc0[r] + bv0;
  }
  if (c1 < CDIM) {
    const float bv1 = b2[c1];
#pragma unroll
    for (int r = 0; r < 8; ++r) {
      const int orow = rowTile + r + sel * 8;
      h[(long)orow * CDIM + c1] = acc1[r] + bv1;
    }
  }
}

// ---------------------------------------------------------------------------
// GAT per-head kernels (banded graph: in-neighbors of j are j-3..j+3 in-doc)
// ---------------------------------------------------------------------------
__global__ void k_zero(float* __restrict__ p, int n) {
  int t = blockIdx.x * blockDim.x + threadIdx.x;
  if (t < n) p[t] = 0.f;
}

__global__ void k_zattr(const float* __restrict__ h, const float* __restrict__ W,
                        const float* __restrict__ as, const float* __restrict__ ad,
                        float* __restrict__ z, float* __restrict__ es,
                        float* __restrict__ ed) {
  int n = blockIdx.x * blockDim.x + threadIdx.x;
  if (n >= NNODES) return;
  float hv[CDIM];
#pragma unroll
  for (int c = 0; c < CDIM; ++c) hv[c] = h[(long)n * CDIM + c];
  float s_ = 0.f, d_ = 0.f;
#pragma unroll
  for (int d = 0; d < CDIM; ++d) {
    float acc = 0.f;
#pragma unroll
    for (int c = 0; c < CDIM; ++c) acc += hv[c] * W[c * CDIM + d];
    z[(long)n * CDIM + d] = acc;
    s_ += acc * as[d];
    d_ += acc * ad[d];
  }
  es[n] = s_;
  ed[n] = d_;
}

__global__ void k_att(const float* __restrict__ es, const float* __restrict__ ed,
                      float* __restrict__ att) {
  int j = blockIdx.x * blockDim.x + threadIdx.x;
  if (j >= NNODES) return;
  const int p = j % LDOC;
  const float edj = ed[j];
  float ev[7];
  bool valid[7];
  float m = -1e30f;
#pragma unroll
  for (int di = -3; di <= 3; ++di) {
    const int q = p + di;
    const int idx = di + 3;
    valid[idx] = (q >= 0) && (q < LDOC);
    if (valid[idx]) {
      float v = es[j + di] + edj;
      v = v > 0.f ? v : 0.2f * v;   // leaky_relu(0.2)
      ev[idx] = v;
      m = v > m ? v : m;
    } else {
      ev[idx] = 0.f;
    }
  }
  float s = 0.f;
  float ex[7];
#pragma unroll
  for (int idx = 0; idx < 7; ++idx) {
    ex[idx] = valid[idx] ? expf(ev[idx] - m) : 0.f;
    s += ex[idx];
  }
  const float inv = 1.f / (s + 1e-9f);
#pragma unroll
  for (int idx = 0; idx < 7; ++idx) att[(long)j * 8 + idx] = ex[idx] * inv;
}

__global__ void k_hop(const float* __restrict__ att, const float* __restrict__ zt_old,
                      const float* __restrict__ z0, float* __restrict__ zt_new) {
  int t = blockIdx.x * blockDim.x + threadIdx.x;
  if (t >= NNODES * CDIM) return;
  const int j = t / CDIM;
  const int c = t - j * CDIM;
  const int p = j % LDOC;
  float agg = 0.f;
#pragma unroll
  for (int di = -3; di <= 3; ++di) {
    const int q = p + di;
    if (q >= 0 && q < LDOC)
      agg += att[(long)j * 8 + (di + 3)] * zt_old[(long)(j + di) * CDIM + c];
  }
  zt_new[t] = (1.f - ALPHA_) * agg + ALPHA_ * z0[t];
}

__global__ void k_accum(const float* __restrict__ zt, float* __restrict__ hout) {
  int t = blockIdx.x * blockDim.x + threadIdx.x;
  if (t >= NNODES * CDIM) return;
  float v = zt[t];
  float e = v > 0.f ? v : (expf(v) - 1.f);
  hout[t] += e * (1.f / NHEADS);
}

__global__ __launch_bounds__(256) void k_final(
    const float* __restrict__ h, const float* __restrict__ wg,
    const float* __restrict__ bg, float* __restrict__ out) {
  __shared__ float acc[CDIM];
  const int doc = blockIdx.x;
  if (threadIdx.x < CDIM) acc[threadIdx.x] = 0.f;
  __syncthreads();
  for (int p = threadIdx.x; p < LDOC; p += blockDim.x) {
    const long n = (long)doc * LDOC + p;
    float g = bg[0];
    float hv[CDIM];
#pragma unroll
    for (int c = 0; c < CDIM; ++c) {
      hv[c] = h[n * CDIM + c];
      g += hv[c] * wg[c];
    }
    g = 1.f / (1.f + expf(-g));
#pragma unroll
    for (int c = 0; c < CDIM; ++c) atomicAdd(&acc[c], g * hv[c]);
  }
  __syncthreads();
  if (threadIdx.x < CDIM) out[doc * CDIM + threadIdx.x] = acc[threadIdx.x];
}

// ---------------------------------------------------------------------------
extern "C" void kernel_launch(void* const* d_in, const int* in_sizes, int n_in,
                              void* d_out, int out_size, void* d_ws, size_t ws_size,
                              hipStream_t stream) {
  (void)in_sizes; (void)n_in; (void)out_size; (void)ws_size;
  const int*   node_ids = (const int*)d_in[0];
  const float* emb    = (const float*)d_in[4];
  const float* W1     = (const float*)d_in[5];
  const float* b1     = (const float*)d_in[6];
  const float* W2     = (const float*)d_in[7];
  const float* b2     = (const float*)d_in[8];
  const float* gat_W  = (const float*)d_in[9];
  const float* a_src  = (const float*)d_in[10];
  const float* a_dst  = (const float*)d_in[11];
  const float* w_gate = (const float*)d_in[12];
  const float* b_gate = (const float*)d_in[13];
  float* out = (float*)d_out;

  char* ws = (char*)d_ws;
  size_t off = 0;
  auto alloc = [&](size_t bytes) {
    char* p = ws + off;
    off = (off + bytes + 255) & ~(size_t)255;
    return p;
  };
  // region 0: Xg (dead after GEMM1) -- later aliased by GAT buffers
  __bf16* Xg  = (__bf16*)alloc((size_t)NNODES * DIM * sizeof(__bf16));   // 68.8 MB
  __bf16* X1  = (__bf16*)alloc((size_t)NNODES * HID * sizeof(__bf16));   // 22.9 MB
  __bf16* W1t = (__bf16*)alloc((size_t)HID * DIM * sizeof(__bf16));
  __bf16* W2t = (__bf16*)alloc((size_t)32 * HID * sizeof(__bf16));
  // GAT buffers alias the Xg region (only live after GEMM2)
  size_t goff = 0;
  auto galloc = [&](size_t bytes) {
    char* p = (char*)Xg + goff;
    goff = (goff + bytes + 255) & ~(size_t)255;
    return p;
  };
  float* h0  = (float*)galloc((size_t)NNODES * CDIM * sizeof(float));
  float* h1  = (float*)galloc((size_t)NNODES * CDIM * sizeof(float));
  float* z   = (float*)galloc((size_t)NNODES * CDIM * sizeof(float));
  float* zt0 = (float*)galloc((size_t)NNODES * CDIM * sizeof(float));
  float* zt1 = (float*)galloc((size_t)NNODES * CDIM * sizeof(float));
  float* es  = (float*)galloc((size_t)NNODES * sizeof(float));
  float* ed  = (float*)galloc((size_t)NNODES * sizeof(float));
  float* att = (float*)galloc((size_t)NNODES * 8 * sizeof(float));

  const int tpbN  = (NNODES + 255) / 256;
  const int tpbNC = (NNODES * CDIM + 255) / 256;

  // one-time conversions (bf16 + transposed weights, gathered bf16 activations)
  k_cvt_w1t<<<(DIM * HID + 255) / 256, 256, 0, stream>>>(W1, W1t);
  k_cvt_w2t<<<(32 * HID + 255) / 256, 256, 0, stream>>>(W2, W2t);
  k_gather_cvt<<<NNODES, 192, 0, stream>>>(node_ids, emb, (unsigned*)Xg);

  // WMMA GEMMs
  k_gemm1<<<NNODES / 16, 128, 0, stream>>>(Xg, W1t, b1, X1);
  k_gemm2<<<NNODES / 64, 128, 0, stream>>>(X1, W2t, b2, h0);

  float* hin = h0;
  float* hout = h1;
  for (int l = 0; l < NLAYERS; ++l) {
    k_zero<<<tpbNC, 256, 0, stream>>>(hout, NNODES * CDIM);
    for (int hd = 0; hd < NHEADS; ++hd) {
      const float* Wh = gat_W + (size_t)(l * NHEADS + hd) * CDIM * CDIM;
      const float* as = a_src + (size_t)(l * NHEADS + hd) * CDIM;
      const float* ad = a_dst + (size_t)(l * NHEADS + hd) * CDIM;
      k_zattr<<<tpbN, 256, 0, stream>>>(hin, Wh, as, ad, z, es, ed);
      k_att<<<tpbN, 256, 0, stream>>>(es, ed, att);
      k_hop<<<tpbNC, 256, 0, stream>>>(att, z,   z, zt0);
      k_hop<<<tpbNC, 256, 0, stream>>>(att, zt0, z, zt1);
      k_hop<<<tpbNC, 256, 0, stream>>>(att, zt1, z, zt0);
      k_accum<<<tpbNC, 256, 0, stream>>>(zt0, hout);
    }
    float* t = hin; hin = hout; hout = t;
  }
  k_final<<<NDOCS, 256, 0, stream>>>(hin, w_gate, b_gate, out);
}
